// MinGRU_7842610283421
// MI455X (gfx1250) — compile-verified, matrix operand order checked by reading, not run on hardware
//
#include <hip/hip_runtime.h>

// MinGRU fused: bf16-WMMA GEMM (hg = x @ W_hg) + log-space epilogue +
// sequential Heinsen scan, single pass over S. B=8, S=4096, D=K=1024.
//
// d_in[0] = x      [8,4096,1024] f32
// d_in[1] = W_hg   [1024,2048]   f32   (cols 0..1023 hidden, 1024..2047 gate)
// d_out   = out    [8,4096,1024] f32
//
// Per block: batch b, 32-channel slab. 4 waves:
//   wave0: channels 0-15,  K   0..511  (h+g)     wave1: channels 16-31, K   0..511
//   wave2: channels 0-15,  K 512..1023 (h+g)     wave3: channels 16-31, K 512..1023
// x tiles (contiguous 64 KB) stream via global_load_async_to_lds_b128 into a
// double-buffered LDS staging area (ASYNCcnt), then get converted to bf16
// WMMA A-fragments. W slab lives in LDS as bf16 B-fragments for the whole run.

typedef __attribute__((ext_vector_type(16))) __bf16       v16bf;
typedef __attribute__((ext_vector_type(8)))  float        v8f;
typedef __attribute__((ext_vector_type(4)))  float        v4f;
typedef __attribute__((ext_vector_type(2)))  unsigned int v2u;

#define S_LEN   4096
#define D_DIM   1024
#define TILE_S  16
#define STILES  (S_LEN / TILE_S)      // 256
#define TD      32                    // channels per block
#define NTHR    128                   // 4 waves

// LDS carve (bytes); dynamic LDS starts at offset 0 (no static __shared__)
#define LDS_W_OFF    0          // 4 frags * 32 kb * 32 lanes * 16 bf16 = 128 KB
#define LDS_XF_OFF   131072     // 32 kb * 32 lanes * 16 bf16           =  32 KB
#define LDS_RAW0_OFF 163840     // 16 x 1024 f32 raw tile               =  64 KB
#define LDS_RAW1_OFF 229376     //                                      =  64 KB
#define LDS_EX_OFF   294912     // 4 x (16 x 32) f32 exchange           =   8 KB
#define LDS_TOTAL    303104

// f32 -> bf16, round-to-nearest-even
static __device__ __forceinline__ unsigned short f2bf(float f) {
  unsigned int u = __float_as_uint(f);
  unsigned int r = u + 0x7FFFu + ((u >> 16) & 1u);
  return (unsigned short)(r >> 16);
}

static __device__ __forceinline__ float softplusf(float v) {
  return fmaxf(v, 0.f) + __logf(1.f + __expf(-fabsf(v)));
}

// Async DMA of one contiguous 64 KB tile (global -> LDS), 32 x b128 per thread.
static __device__ __forceinline__ void async_copy_tile(const float* gbase,
                                                       unsigned lds_off, int tid) {
  unsigned long long sa = (unsigned long long)gbase;
  #pragma unroll
  for (int q = 0; q < 32; ++q) {
    unsigned off = ((unsigned)(q * NTHR + tid)) << 4;   // byte offset, 16B chunks
    asm volatile("global_load_async_to_lds_b128 %0, %1, %2"
                 :: "v"(lds_off + off), "v"(off), "s"(sa)
                 : "memory");
  }
}

// ISA 7.12.2 16-bit fragment mapping: (row, kk) -> lane = half*16+row, slot i
//   half = (kk>>3)&1 ;  i = (kk&7) + ((kk>>4)<<3)

__global__ __launch_bounds__(NTHR)
void mingru_fused_kernel(const float* __restrict__ x,
                         const float* __restrict__ W,
                         float* __restrict__ out) {
  extern __shared__ char smem[];
  unsigned short* lds_w  = (unsigned short*)(smem + LDS_W_OFF);
  unsigned short* lds_xf = (unsigned short*)(smem + LDS_XF_OFF);
  float*          lds_r0 = (float*)(smem + LDS_RAW0_OFF);
  float*          lds_r1 = (float*)(smem + LDS_RAW1_OFF);
  float*          lds_h1 = (float*)(smem + LDS_EX_OFF);
  float*          lds_g1 = lds_h1 + 512;
  float*          lds_h2 = lds_g1 + 512;
  float*          lds_g2 = lds_h2 + 512;

  const int tid  = threadIdx.x;
  const int lane = tid & 31;
  const int wave = tid >> 5;
  const int b    = blockIdx.x >> 5;                // 32 d-slabs per batch
  const int d0   = (blockIdx.x & 31) * TD;

  // ---- one-time: W slab (64 cols x 1024 K) -> bf16 B-fragments in LDS
  // frag f: 0 = hidden cols 0-15, 1 = hidden 16-31, 2 = gate 0-15, 3 = gate 16-31
  for (int it = tid; it < 64 * 256; it += NTHR) {
    int c   = it & 63;
    int k4  = (it >> 6) << 2;
    int col = (c < TD) ? (d0 + c) : (1024 + d0 + (c - TD));
    const float* wp = W + (size_t)k4 * 2048 + col;
    unsigned int lo = (unsigned int)f2bf(wp[0])      | ((unsigned int)f2bf(wp[2048]) << 16);
    unsigned int hi = (unsigned int)f2bf(wp[2*2048]) | ((unsigned int)f2bf(wp[3*2048]) << 16);
    int kb = k4 >> 5, kk = k4 & 31;
    int half = (kk >> 3) & 1;
    int i0   = (kk & 7) + ((kk >> 4) << 3);
    int ln   = (half << 4) + (c & 15);
    v2u pk = {lo, hi};
    *(v2u*)(lds_w + (((((size_t)(c >> 4)) * 32 + kb) * 32 + ln) << 4) + i0) = pk;
  }

  const float* xb = x   + (size_t)b * S_LEN * D_DIM;
  float*       ob = out + (size_t)b * S_LEN * D_DIM + d0;

  // per-channel scan state (lanes of wave 0), lives in VGPRs across all of S
  float a_run = 0.f;
  float mx    = -__builtin_inff();
  float sm    = 0.f;

  // prologue: start DMA of tile 0
  async_copy_tile(xb, LDS_RAW0_OFF, tid);

  for (int st = 0; st < STILES; ++st) {
    const int s0  = st << 4;
    const int cur = st & 1;

    // ---- kick DMA of next tile into the other raw buffer
    if (st + 1 < STILES) {
      async_copy_tile(xb + (size_t)(s0 + TILE_S) * D_DIM,
                      cur ? LDS_RAW0_OFF : LDS_RAW1_OFF, tid);
      asm volatile("s_wait_asynccnt 0x20" ::: "memory");  // tile st complete
    } else {
      asm volatile("s_wait_asynccnt 0x0" ::: "memory");
    }
    __syncthreads();

    // ---- convert raw f32 tile -> bf16 A-fragments (LDS -> LDS)
    {
      const float* raw = cur ? lds_r1 : lds_r0;
      for (int it = tid; it < 4096; it += NTHR) {
        int k4 = (it & 255) << 2;
        int m  = it >> 8;
        v4f f = *(const v4f*)(raw + m * D_DIM + k4);
        int kb = k4 >> 5, kk = k4 & 31;
        int half = (kk >> 3) & 1;
        int i0   = (kk & 7) + ((kk >> 4) << 3);
        int ln   = (half << 4) + m;
        unsigned int lo = (unsigned int)f2bf(f.x) | ((unsigned int)f2bf(f.y) << 16);
        unsigned int hi = (unsigned int)f2bf(f.z) | ((unsigned int)f2bf(f.w) << 16);
        v2u pk = {lo, hi};
        *(v2u*)(lds_xf + (((kb * 32 + ln) << 4) + i0)) = pk;
      }
    }
    __syncthreads();

    // ---- GEMM: each wave -> 16x16 h-tile AND g-tile over its K half
    //      one A fragment feeds two WMMAs (h and g)
    v8f acc_h = {}, acc_g = {};
    {
      const int kb0 = (wave >> 1) << 4;                                  // K half
      const unsigned short* bwh = lds_w + ((size_t)(wave & 1) << 14);
      const unsigned short* bwg = lds_w + ((size_t)(2 + (wave & 1)) << 14);
      #pragma unroll 4
      for (int kk = 0; kk < 16; ++kk) {
        int kb = kb0 + kk;
        v16bf af = *(const v16bf*)(lds_xf + ((kb * 32 + lane) << 4));
        v16bf bh = *(const v16bf*)(bwh    + ((kb * 32 + lane) << 4));
        v16bf bg = *(const v16bf*)(bwg    + ((kb * 32 + lane) << 4));
        acc_h = __builtin_amdgcn_wmma_f32_16x16x32_bf16(
            false, af, false, bh, (short)0, acc_h, false, false);
        acc_g = __builtin_amdgcn_wmma_f32_16x16x32_bf16(
            false, af, false, bg, (short)0, acc_g, false, false);
      }
    }

    // ---- K-partials -> exchange buffers (C layout: vgpr j -> M=j or j+8)
    {
      int m0 = (lane >> 4) << 3;
      int nn = ((wave & 1) << 4) + (lane & 15);
      float* dh = (wave < 2) ? lds_h1 : lds_h2;
      float* dg = (wave < 2) ? lds_g1 : lds_g2;
      #pragma unroll
      for (int j = 0; j < 8; ++j) {
        dh[((m0 + j) << 5) + nn] = acc_h[j];
        dg[((m0 + j) << 5) + nn] = acc_g[j];
      }
    }
    __syncthreads();

    // ---- reduce K halves + log-space elementwise terms (all threads)
    for (int idx = tid; idx < TILE_S * TD; idx += NTHR) {
      float g = lds_g1[idx] + lds_g2[idx];
      float h = lds_h1[idx] + lds_h2[idx];
      float lc  = -softplusf(g);                                   // log(1-z)
      float lz  = -softplusf(-g);                                  // log z
      float lgh = (h >= 0.f) ? __logf(h + 0.5f) : -softplusf(-h);  // log g(h)
      lds_g1[idx] = lc;
      lds_h1[idx] = lz + lgh;                                      // log_values
    }
    __syncthreads();

    // ---- sequential scan: one channel per lane, state in registers
    if (tid < TD) {
      #pragma unroll
      for (int m = 0; m < TILE_S; ++m) {
        a_run += lds_g1[(m << 5) + tid];                // a* = cumsum(log_coeffs)
        float t = lds_h1[(m << 5) + tid] - a_run;       // log_values - a*
        if (t > mx) { sm = sm * __expf(mx - t) + 1.f; mx = t; }   // online LSE
        else        { sm += __expf(t - mx); }
        ob[(size_t)(s0 + m) * D_DIM + tid] = __expf(a_run + mx) * sm;
      }
    }
  }
}

extern "C" void kernel_launch(void* const* d_in, const int* in_sizes, int n_in,
                              void* d_out, int out_size, void* d_ws, size_t ws_size,
                              hipStream_t stream) {
  (void)in_sizes; (void)n_in; (void)out_size; (void)d_ws; (void)ws_size;
  const float* x = (const float*)d_in[0];
  const float* W = (const float*)d_in[1];
  float* out     = (float*)d_out;

  dim3 grid(8 * (D_DIM / TD));   // 256 blocks: (batch, 32-channel slab)
  dim3 block(NTHR);
  mingru_fused_kernel<<<grid, block, LDS_TOTAL, stream>>>(x, W, out);
}
